// COGV1_17042430230893
// MI455X (gfx1250) — matrix operand order repaired and checked by reading, not compiled, not run on hardware
//
#include <hip/hip_runtime.h>

#define IMG 224

typedef __attribute__((ext_vector_type(16))) __bf16       v16bf;
typedef __attribute__((ext_vector_type(8)))  float        v8f;
typedef __attribute__((ext_vector_type(4)))  unsigned int u32x4;
typedef __attribute__((ext_vector_type(2)))  unsigned int u32x2;

__device__ __forceinline__ unsigned short f2bf(float f) {
  unsigned int u = __float_as_uint(f);
  u += 0x7FFFu + ((u >> 16) & 1u);          // round-to-nearest-even
  return (unsigned short)(u >> 16);
}
__device__ __forceinline__ float bf2f(unsigned short h) {
  return __uint_as_float(((unsigned int)h) << 16);
}

// ---------------------------------------------------------------------------
// Stage 0: pack conv2 weights (OIHW fp32) into exact WMMA B-fragment layout.
// wp[((tap*4+q)*8+ob)*32+lane][e] (bf16): col o = ob*16+lane%16,
// K (within 32-chunk) = (e>>3)*16 + 8*(lane>=16) + (e&7)  ->  cin = q*32 + (e>>3)*16 + 8*hi + (e&7).
// ---------------------------------------------------------------------------
__global__ void k_pack_w2(const float* __restrict__ w2, unsigned short* __restrict__ wp) {
  int idx = blockIdx.x * blockDim.x + threadIdx.x;
  if (idx >= 147456) return;
  int e = idx & 15;
  int r = idx >> 4;
  int L = r & 31;  r >>= 5;
  int ob = r & 7;  r >>= 3;
  int q  = r & 3;
  int tap = r >> 2;
  int ky = tap / 3, kx = tap % 3;
  int o   = ob * 16 + (L & 15);
  int cin = q * 32 + (e >> 3) * 16 + 8 * (L >> 4) + (e & 7);
  wp[idx] = f2bf(w2[((o * 128 + cin) * 3 + ky) * 3 + kx]);
}

// ---------------------------------------------------------------------------
// Stage 1a: windowed antialias (triangle) resize, zero-pad handled inline.
// dst layout: (b, ci) slabs of 2048 fp32, row-major (sh, sw).
// ---------------------------------------------------------------------------
__global__ void k_resize_win(const float* __restrict__ inp, float* __restrict__ dst,
                             int t, int l, int Hin, int Wn, int sh, int sw) {
  int idx = blockIdx.x * blockDim.x + threadIdx.x;
  int total = 16 * 3 * sh * sw;
  if (idx >= total) return;
  int x = idx % sw;
  int y = (idx / sw) % sh;
  int ci = (idx / (sw * sh)) % 3;
  int b  = idx / (sw * sh * 3);
  float ry = (float)sh / (float)Hin;
  float rx = (float)sw / (float)Wn;
  float cy = (y + 0.5f) / ry - 0.5f;
  float cx = (x + 0.5f) / rx - 0.5f;
  float supy = 1.0f / ry, supx = 1.0f / rx;
  int jy0 = max(0, (int)ceilf(cy - supy)), jy1 = min(Hin - 1, (int)floorf(cy + supy));
  int jx0 = max(0, (int)ceilf(cx - supx)), jx1 = min(Wn  - 1, (int)floorf(cx + supx));
  float acc = 0.f, wsum = 0.f;
  for (int jy = jy0; jy <= jy1; ++jy) {
    float wy = 1.0f - fabsf((jy - cy) * ry);
    if (wy <= 0.f) continue;
    int gy = t + jy - 6;                       // window coords -> image coords (PAD=6)
    for (int jx = jx0; jx <= jx1; ++jx) {
      float wx = 1.0f - fabsf((jx - cx) * rx);
      if (wx <= 0.f) continue;
      float wgt = wy * wx;
      wsum += wgt;
      int gx = l + jx - 6;
      float v = 0.f;
      if (gy >= 0 && gy < IMG && gx >= 0 && gx < IMG)
        v = inp[((b * 3 + ci) * IMG + gy) * IMG + gx];
      acc += wgt * v;
    }
  }
  dst[(b * 3 + ci) * 2048 + y * sw + x] = acc / wsum;
}

// ---------------------------------------------------------------------------
// Stage 1b: direct 7x7 VALID conv (3->128). Weights staged in LDS (bf16,
// [i][c] layout -> conflict-free), patch staged in LDS (broadcast reads).
// Output slab layout: (b*1024 + pix)*128 + c (channel-contiguous).
// ---------------------------------------------------------------------------
__global__ __launch_bounds__(256) void k_conv1_win(const float* __restrict__ xd,
    const float* __restrict__ w1, float* __restrict__ slab,
    int sh, int sw, int fh, int fw) {
  __shared__ unsigned short wlds[147 * 128];
  __shared__ float patch[2][148];
  const int npix = fh * fw;
  const int b = blockIdx.y;
  for (int i = threadIdx.x; i < 147 * 128; i += 256) {
    int wi = i >> 7, c = i & 127;
    int ci = wi / 49, rr = wi % 49, ky = rr / 7, kx = rr % 7;
    wlds[i] = f2bf(w1[((c * 3 + ci) * 7 + ky) * 7 + kx]);
  }
  const int c    = threadIdx.x & 127;
  const int slot = threadIdx.x >> 7;
  const int p0   = blockIdx.x * 128;
  const int pend = min(p0 + 128, npix);
  for (int pb = p0; pb < pend; pb += 2) {
    __syncthreads();
    for (int i = threadIdx.x; i < 2 * 147; i += 256) {
      int sl = i / 147, ii = i % 147;
      int pix = pb + sl;
      float v = 0.f;
      if (pix < npix) {
        int fy = pix / fw, fx = pix % fw;
        int ci = ii / 49, rr = ii % 49, ky = rr / 7, kx = rr % 7;
        v = xd[(b * 3 + ci) * 2048 + (fy + ky) * sw + (fx + kx)];
      }
      patch[sl][ii] = v;
    }
    __syncthreads();
    int pix = pb + slot;
    if (pix < pend) {
      float acc = 0.f;
#pragma unroll 7
      for (int i = 0; i < 147; ++i)
        acc = fmaf(patch[slot][i], bf2f(wlds[i * 128 + c]), acc);
      slab[(b * 1024 + pix) * 128 + c] = acc;
    }
  }
}

// ---------------------------------------------------------------------------
// Stage 1c: nearest upsample of conv1 window into full map (NHWC bf16).
// ---------------------------------------------------------------------------
__global__ void k_upsample_win(const float* __restrict__ slab, unsigned short* __restrict__ out1,
                               int t, int l, int h, int w, int fh, int fw) {
  int idx = blockIdx.x * blockDim.x + threadIdx.x;
  int total = 16 * h * w * 128;
  if (idx >= total) return;
  int c = idx & 127;
  int r = idx >> 7;
  int xl = r % w;  r /= w;
  int yl = r % h;
  int b  = r / h;
  int sy = (yl * fh) / h, sx = (xl * fw) / w;
  float v = slab[(b * 1024 + sy * fw + sx) * 128 + c];
  out1[((b * IMG + (t + yl)) * IMG + (l + xl)) * 128 + c] = f2bf(v);
}

// ---------------------------------------------------------------------------
// BN statistics, deterministic two-stage reduction (no atomics).
// ---------------------------------------------------------------------------
__global__ __launch_bounds__(256) void k_bn_stats_bf16(const unsigned short* __restrict__ x,
    float* __restrict__ psum, float* __restrict__ pss, int S, int perblk) {
  __shared__ float ssum[256];
  __shared__ float sss[256];
  int c = threadIdx.x & 127, g = threadIdx.x >> 7;
  int start = blockIdx.x * perblk;
  int end = min(start + perblk, S);
  float s = 0.f, q = 0.f;
  for (int p = start + g; p < end; p += 2) {
    float v = bf2f(x[p * 128 + c]);
    s += v;
    q = fmaf(v, v, q);
  }
  ssum[threadIdx.x] = s;
  sss[threadIdx.x]  = q;
  __syncthreads();
  if (g == 0) {
    psum[blockIdx.x * 128 + c] = ssum[c] + ssum[c + 128];
    pss [blockIdx.x * 128 + c] = sss[c]  + sss[c + 128];
  }
}

__global__ __launch_bounds__(256) void k_bn_stats_f32(const float* __restrict__ x,
    float* __restrict__ psum, float* __restrict__ pss, int S, int perblk) {
  __shared__ float ssum[256];
  __shared__ float sss[256];
  int c = threadIdx.x & 127, g = threadIdx.x >> 7;
  int start = blockIdx.x * perblk;
  int end = min(start + perblk, S);
  float s = 0.f, q = 0.f;
  for (int p = start + g; p < end; p += 2) {
    float v = x[p * 128 + c];
    s += v;
    q = fmaf(v, v, q);
  }
  ssum[threadIdx.x] = s;
  sss[threadIdx.x]  = q;
  __syncthreads();
  if (g == 0) {
    psum[blockIdx.x * 128 + c] = ssum[c] + ssum[c + 128];
    pss [blockIdx.x * 128 + c] = sss[c]  + sss[c + 128];
  }
}

__global__ void k_bn_finalize(const float* __restrict__ psum, const float* __restrict__ pss,
                              const float* __restrict__ gamma, const float* __restrict__ beta,
                              float* __restrict__ ab, int nblk, float invN) {
  int c = threadIdx.x;  // 128 threads
  float s = 0.f, q = 0.f;
  for (int i = 0; i < nblk; ++i) { s += psum[i * 128 + c]; q += pss[i * 128 + c]; }
  float mean = s * invN;
  float var  = q * invN - mean * mean;
  float a = gamma[c] * rsqrtf(var + 1e-5f);
  ab[c]       = a;
  ab[128 + c] = beta[c] - mean * a;
}

// ---------------------------------------------------------------------------
// Stage 2: fused BN1-affine + ReLU + 3x3/2 maxpool (pad 1).  NHWC bf16 out.
// ---------------------------------------------------------------------------
__global__ void k_bnrelu_pool(const unsigned short* __restrict__ out1,
                              const float* __restrict__ ab,
                              unsigned short* __restrict__ pooled) {
  int idx = blockIdx.x * blockDim.x + threadIdx.x;
  if (idx >= 16 * 112 * 112 * 128) return;
  int c = idx & 127;
  int r = idx >> 7;
  int x = r % 112; r /= 112;
  int y = r % 112;
  int b = r / 112;
  float a = ab[c], bb = ab[128 + c];
  float m = 0.f;  // relu(...) >= 0, window always non-empty
  for (int ky = 0; ky < 3; ++ky) {
    int yy = 2 * y - 1 + ky;
    if (yy < 0 || yy >= IMG) continue;
    for (int kx = 0; kx < 3; ++kx) {
      int xx = 2 * x - 1 + kx;
      if (xx < 0 || xx >= IMG) continue;
      float v = bf2f(out1[((b * IMG + yy) * IMG + xx) * 128 + c]);
      m = fmaxf(m, fmaxf(fmaf(v, a, bb), 0.f));
    }
  }
  pooled[idx] = f2bf(m);
}

// ---------------------------------------------------------------------------
// Stage 3: 3x3 conv 128->128 via WMMA bf16 implicit GEMM.
// Block = 256 threads (8 waves); block tile = 16 x-positions x 4 y-rows x all
// 128 output channels.  Wave w owns out-channels [16w,16w+16) and keeps FOUR
// 16x16 accumulators (rows y0..y0+3): each B (weight) fragment feeds 4 WMMAs.
// All four A fragments are loaded into distinct registers BEFORE the 4 WMMAs
// and the whole K-loop (9 taps x 4 chunks) is fully unrolled so the scheduler
// can cluster the ds/global loads and hoist the next group above the current
// WMMAs (double-buffering instead of per-WMMA s_wait_dscnt 0).
// LDS: pooled[b, y0-1..y0+4, x0-1..x0+16, :] bf16, channel stride padded to
// 136 (272 B -> per-lane bank offset 4 words, conflict-free ds_load_b128).
// ---------------------------------------------------------------------------
__global__ __launch_bounds__(256) void k_conv2_wmma(const unsigned short* __restrict__ pooled,
    const unsigned short* __restrict__ wp, float* __restrict__ out2) {
  __shared__ __attribute__((aligned(16))) unsigned short tile[6 * 18 * 136];  // 29376 B
  const int x0 = blockIdx.x * 16;   // 7 tiles of 16 -> 112
  const int y0 = blockIdx.y * 4;    // 28 tiles of 4 rows -> 112
  const int b  = blockIdx.z;        // 16

  // --- cooperative LDS fill: 6 input rows (zero-padded halo) ---
  for (int i = threadIdx.x; i < 108 * 32; i += 256) {
    int row = i >> 5, seg = i & 31;
    int dy = row / 18, xx = row % 18;
    int gy = y0 - 1 + dy, gx = x0 - 1 + xx;
    u32x2 val = (u32x2)0u;
    if (gy >= 0 && gy < 112 && gx >= 0 && gx < 112)
      val = *(const u32x2*)(pooled + (((b * 112 + gy) * 112 + gx) * 128 + seg * 4));
    *(u32x2*)(tile + (dy * 18 + xx) * 136 + seg * 4) = val;
  }
  __syncthreads();

  const int wv   = threadIdx.x >> 5;   // wave 0..7 -> out-channel tile
  const int lane = threadIdx.x & 31;
  const int m16  = lane & 15;          // A row = spatial x offset
  const int hi   = lane >> 4;
  const int o0   = wv * 16;

  union Frag { v16bf v; u32x4 q[2]; };
  v8f acc0 = (v8f){0.f, 0.f, 0.f, 0.f, 0.f, 0.f, 0.f, 0.f};
  v8f acc1 = acc0, acc2 = acc0, acc3 = acc0;

#pragma unroll
  for (int tap = 0; tap < 9; ++tap) {
    const int dyi = tap / 3, dxi = tap % 3;
    const unsigned short* wbase = wp + (((tap * 4) * 8 + wv) * 32 + lane) * 16;
    __builtin_prefetch(wbase + 4096, 0, 1);   // next q-chunk's weights (same tap) -> global_prefetch
#pragma unroll
    for (int q = 0; q < 4; ++q) {
      // B: pre-packed, 32 contiguous bytes per lane; reused by 4 row-WMMAs.
      Frag B, A0, A1, A2, A3;
      const unsigned short* wb = wbase + q * (8 * 32 * 16);
      B.q[0] = *(const u32x4*)(wb);
      B.q[1] = *(const u32x4*)(wb + 8);
      // A: lane holds row m16; element e -> chan = q*32 + (e>>3)*16 + hi*8 + (e&7)
      const int offA = (dyi * 18 + (m16 + dxi)) * 136 + q * 32 + hi * 8;
      A0.q[0] = *(const u32x4*)(tile + offA);
      A0.q[1] = *(const u32x4*)(tile + offA + 16);
      A1.q[0] = *(const u32x4*)(tile + offA + 18 * 136);
      A1.q[1] = *(const u32x4*)(tile + offA + 18 * 136 + 16);
      A2.q[0] = *(const u32x4*)(tile + offA + 36 * 136);
      A2.q[1] = *(const u32x4*)(tile + offA + 36 * 136 + 16);
      A3.q[0] = *(const u32x4*)(tile + offA + 54 * 136);
      A3.q[1] = *(const u32x4*)(tile + offA + 54 * 136 + 16);
      acc0 = __builtin_amdgcn_wmma_f32_16x16x32_bf16(false, A0.v, false, B.v, (short)0, acc0, false, false);
      acc1 = __builtin_amdgcn_wmma_f32_16x16x32_bf16(false, A1.v, false, B.v, (short)0, acc1, false, false);
      acc2 = __builtin_amdgcn_wmma_f32_16x16x32_bf16(false, A2.v, false, B.v, (short)0, acc2, false, false);
      acc3 = __builtin_amdgcn_wmma_f32_16x16x32_bf16(false, A3.v, false, B.v, (short)0, acc3, false, false);
    }
  }

  // D layout: VGPR v -> row v + 8*hi, col = m16
  {
    float* outp = out2 + ((b * 112 + y0) * 112) * 128;
#pragma unroll
    for (int v = 0; v < 8; ++v) outp[(x0 + v + 8 * hi) * 128 + o0 + m16] = acc0[v];
    outp += 112 * 128;
#pragma unroll
    for (int v = 0; v < 8; ++v) outp[(x0 + v + 8 * hi) * 128 + o0 + m16] = acc1[v];
    outp += 112 * 128;
#pragma unroll
    for (int v = 0; v < 8; ++v) outp[(x0 + v + 8 * hi) * 128 + o0 + m16] = acc2[v];
    outp += 112 * 128;
#pragma unroll
    for (int v = 0; v < 8; ++v) outp[(x0 + v + 8 * hi) * 128 + o0 + m16] = acc3[v];
  }
}

// ---------------------------------------------------------------------------
// Stage 4: BN2 affine + ReLU, NHWC fp32 -> NCHW fp32 output.
// ---------------------------------------------------------------------------
__global__ void k_bn2_relu_out(const float* __restrict__ out2, const float* __restrict__ ab,
                               float* __restrict__ out) {
  int idx = blockIdx.x * blockDim.x + threadIdx.x;
  if (idx >= 16 * 128 * 112 * 112) return;
  int x = idx % 112;
  int r = idx / 112;
  int y = r % 112; r /= 112;
  int c = r % 128;
  int b = r / 128;
  float v = out2[((b * 112 + y) * 112 + x) * 128 + c];
  out[idx] = fmaxf(fmaf(v, ab[c], ab[128 + c]), 0.f);
}

// ---------------------------------------------------------------------------
extern "C" void kernel_launch(void* const* d_in, const int* in_sizes, int n_in,
                              void* d_out, int out_size, void* d_ws, size_t ws_size,
                              hipStream_t stream) {
  (void)in_sizes; (void)n_in; (void)out_size; (void)ws_size;
  const float* inp = (const float*)d_in[0];
  const float* w1  = (const float*)d_in[1];
  const float* g1  = (const float*)d_in[2];
  const float* be1 = (const float*)d_in[3];
  const float* w2  = (const float*)d_in[4];
  const float* g2  = (const float*)d_in[5];
  const float* be2 = (const float*)d_in[6];
  float* out = (float*)d_out;

  char* base = (char*)d_ws;
  size_t off = 0;
  auto carve = [&](size_t bytes) -> char* {
    char* p = base + off;
    off += (bytes + 255) & ~(size_t)255;
    return p;
  };
  unsigned short* out1   = (unsigned short*)carve((size_t)16 * 224 * 224 * 128 * 2);
  unsigned short* pooled = (unsigned short*)carve((size_t)16 * 112 * 112 * 128 * 2);
  float*          out2   = (float*)         carve((size_t)16 * 112 * 112 * 128 * 4);
  unsigned short* wp     = (unsigned short*)carve((size_t)147456 * 2);
  float*          slabR  = (float*)         carve((size_t)16 * 3 * 2048 * 4);
  float*          slabC  = (float*)         carve((size_t)16 * 1024 * 128 * 4);
  float* p1sum = (float*)carve(512 * 128 * 4);
  float* p1ss  = (float*)carve(512 * 128 * 4);
  float* ab1   = (float*)carve(256 * 4);
  float* p2sum = (float*)carve(512 * 128 * 4);
  float* p2ss  = (float*)carve(512 * 128 * 4);
  float* ab2   = (float*)carve(256 * 4);

  k_pack_w2<<<(147456 + 255) / 256, 256, 0, stream>>>(w2, wp);

  // Replicate _windows(): borders = 16*i; scales = fp32(2 - s/6).
  for (int s = 0; s < 7; ++s) {
    float scale = (float)(2.0 - (double)s / 6.0);
    int a = 16 * s, bb = 16 * (s + 1), c = IMG - bb, d = IMG - a;
    int T[4] = {a, bb, c, a}, L[4] = {a, a, bb, c};
    int BO[4] = {bb, d, d, c}, R[4] = {c, bb, d, d};
    for (int k = 0; k < 4; ++k) {
      int t = T[k], l = L[k], bo = BO[k], r = R[k];
      int h = bo - t, w = r - l;
      int Hin = h + 12, Wn = w + 12;
      int sh = (int)((float)Hin / scale);
      int sw = (int)((float)Wn / scale);
      int fh = sh - 12, fw = sw - 12;
      int npix = fh * fw;
      int total_r = 16 * 3 * sh * sw;
      k_resize_win<<<(total_r + 255) / 256, 256, 0, stream>>>(inp, slabR, t, l, Hin, Wn, sh, sw);
      dim3 gc1((npix + 127) / 128, 16);
      k_conv1_win<<<gc1, 256, 0, stream>>>(slabR, w1, slabC, sh, sw, fh, fw);
      int total_u = 16 * h * w * 128;
      k_upsample_win<<<(total_u + 255) / 256, 256, 0, stream>>>(slabC, out1, t, l, h, w, fh, fw);
    }
  }

  k_bn_stats_bf16<<<512, 256, 0, stream>>>(out1, p1sum, p1ss, 802816, 1568);
  k_bn_finalize<<<1, 128, 0, stream>>>(p1sum, p1ss, g1, be1, ab1, 512, 1.0f / 802816.0f);

  {
    int total = 16 * 112 * 112 * 128;
    k_bnrelu_pool<<<(total + 255) / 256, 256, 0, stream>>>(out1, ab1, pooled);
  }

  k_conv2_wmma<<<dim3(7, 28, 16), 256, 0, stream>>>(pooled, wp, out2);

  k_bn_stats_f32<<<512, 256, 0, stream>>>(out2, p2sum, p2ss, 200704, 392);
  k_bn_finalize<<<1, 128, 0, stream>>>(p2sum, p2ss, g2, be2, ab2, 512, 1.0f / 200704.0f);

  {
    int total = 16 * 128 * 112 * 112;
    k_bn2_relu_out<<<(total + 255) / 256, 256, 0, stream>>>(out2, ab2, out);
  }
}
